// GINEEncoder_48962627174811
// MI455X (gfx1250) — compile-verified, matrix operand order checked by reading, not run on hardware
//
#include <hip/hip_runtime.h>

#define N_NODES 50000
#define N_EDGES 800000
#define D       128
#define EDIM    32
#define N_LAYERS 4
#define BN_EPS  1e-5f

typedef __attribute__((ext_vector_type(16))) __bf16       v16bf;
typedef __attribute__((ext_vector_type(8)))  float        v8f;
typedef __attribute__((ext_vector_type(4)))  unsigned int v4u;
typedef __attribute__((ext_vector_type(8)))  int          v8i;
typedef __attribute__((ext_vector_type(4)))  int          v4i;

__device__ __forceinline__ v8f wmma_bf16(v16bf a, v16bf b, v8f c) {
    // D = A(16x32) * B(32x16) + C, f32 accumulate
    return __builtin_amdgcn_wmma_f32_16x16x32_bf16(
        /*neg_a=*/false, a, /*neg_b=*/false, b,
        /*c_mod=*/(short)0, c, /*reuse_a=*/false, /*reuse_b=*/false);
}

// -------- Tensor Data Mover: DMA a rows x cols bf16 tile (row stride = cols)
// from global into LDS, inserting 4 DWORDs (8 bf16) of padding after every
// 64 DWORDs (128 bf16) to match the [rows][cols+8] LDS layout.
// Issue from ONE wave only; follows with s_wait_tensorcnt inside.
__device__ __forceinline__ void tdm_load_bf16_2d(const void* gsrc,
                                                 unsigned lds_off,
                                                 int rows, int cols) {
    unsigned long long ga = (unsigned long long)gsrc;
    v4u g0;
    g0[0] = 1u;                                     // count=1, user mode
    g0[1] = lds_off;                                // lds_addr (bytes)
    g0[2] = (unsigned)(ga & 0xffffffffu);           // global_addr[31:0]
    g0[3] = (unsigned)((ga >> 32) & 0x1ffffffu)     // global_addr[56:32]
          | (2u << 30);                             // type = 2 ("image")
    v8i g1;
    g1[0] = (1 << 16)                               // data_size = 1 (2 bytes)
          | (1 << 20)                               // pad_enable
          | (5 << 22)                               // pad_interval: 64 DWORDs
          | (3 << 25);                              // pad_amount: 4 DWORDs
    g1[1] = (cols & 0xffff) << 16;                  // tensor_dim0[15:0]
    g1[2] = (rows & 0xffff) << 16;                  // tensor_dim1[15:0]
    g1[3] = (cols & 0xffff) << 16;                  // tile_dim0
    g1[4] = (rows & 0xffff);                        // tile_dim1 (tile_dim2=0)
    g1[5] = cols;                                   // tensor_dim0_stride[31:0]
    g1[6] = 0;
    g1[7] = 0;
    v4i z4 = {0, 0, 0, 0};
#if __clang_major__ >= 23
    v8i z8 = {0, 0, 0, 0, 0, 0, 0, 0};
    __builtin_amdgcn_tensor_load_to_lds(g0, g1, z4, z4, z8, 0);
#else
    __builtin_amdgcn_tensor_load_to_lds(g0, g1, z4, z4, 0);
#endif
    __builtin_amdgcn_s_wait_tensorcnt(0);
}

// ---------------------------------------------------------------- utilities
__global__ void zero_f32(float4* __restrict__ p, int n4) {
    int i = blockIdx.x * blockDim.x + threadIdx.x;
    int stride = gridDim.x * blockDim.x;
    const float4 z = {0.f, 0.f, 0.f, 0.f};
    for (; i < n4; i += stride) p[i] = z;
}

__global__ void f32_to_bf16(const float* __restrict__ s, __bf16* __restrict__ d, int n) {
    int i = blockIdx.x * blockDim.x + threadIdx.x;
    int stride = gridDim.x * blockDim.x;
    for (; i < n; i += stride) d[i] = (__bf16)s[i];
}

// ------------------------------------------------- fused edge message kernel
// e = edge_attr @ W_edge + b_edge ; m = relu(x[src] + e) ; agg[dst] += m
// 128 edges per block; 8 waves each own a 16-wide column tile of D=128.
__global__ __launch_bounds__(256)
void gine_edge(const float* __restrict__ x,
               const float* __restrict__ edge_attr,
               const long long* __restrict__ eidx,   // [2][N_EDGES]
               const __bf16* __restrict__ Wedge,     // [EDIM][D] layer slice (bf16)
               const float* __restrict__ bedge,      // [D] layer slice
               float* __restrict__ agg) {
    __shared__ __bf16   sW[EDIM][D + 8];
    __shared__ unsigned sSrcOff[128];                // src * D * 4 (byte offsets)
    __shared__ unsigned sDstOff[128];                // dst * D * 4

    const int t     = threadIdx.x;
    const int lane  = t & 31;
    const int wave  = t >> 5;
    const int ebase = blockIdx.x * 128;

    if (wave == 0)                                   // one TDM issue per block
        tdm_load_bf16_2d(Wedge, (unsigned)(size_t)&sW[0][0], EDIM, D);
    if (t < 128) sSrcOff[t] = (unsigned)(int)eidx[ebase + t] * (unsigned)(D * 4);
    else sDstOff[t - 128] = (unsigned)(int)eidx[N_EDGES + ebase + (t - 128)] * (unsigned)(D * 4);
    __syncthreads();

    const int col16  = lane & 15;
    const int kbase  = (lane < 16) ? 0 : 8;
    const int col    = wave * 16 + col16;
    const int rowoff = (lane < 16) ? 0 : 8;
    const unsigned colB = (unsigned)col * 4u;

    // B operand (W_edge column tile) is invariant across the 8 edge tiles.
    v16bf b;
#pragma unroll
    for (int j = 0; j < 8; ++j) {
        b[j]     = sW[kbase + j][col];
        b[8 + j] = sW[kbase + 16 + j][col];
    }
    const float be = bedge[col];

    for (int et = 0; et < 8; ++et) {
        const int erow = et * 16 + col16;            // A-matrix row (M = lane&15)
        const float* ea = edge_attr + (long long)(ebase + erow) * EDIM;
        __builtin_prefetch(ea + 16 * EDIM, 0, 1);    // next edge tile (streaming)
        v16bf a;
#pragma unroll
        for (int j = 0; j < 8; ++j) {
            a[j]     = (__bf16)ea[kbase + j];
            a[8 + j] = (__bf16)ea[kbase + 16 + j];
        }
        v8f c = {};
        c = wmma_bf16(a, b, c);                      // 16 edges x 16 cols, K=32

        // batch the 8 gathers (saddr + u32 voffset form), then the 8 atomics
        float xv[8];
#pragma unroll
        for (int r = 0; r < 8; ++r) {
            const int el = et * 16 + rowoff + r;     // C row = vgpr + 8*half
            xv[r] = *(const float*)((const char*)x + (size_t)(sSrcOff[el] + colB));
        }
#pragma unroll
        for (int r = 0; r < 8; ++r) {
            const int el = et * 16 + rowoff + r;
            float m = fmaxf(c[r] + be + xv[r], 0.f);
            atomicAdd((float*)((char*)agg + (size_t)(sDstOff[el] + colB)), m);
        }
    }
}

// ------------------------------------------------------- fused node MLP kernel
// h = x + agg ; h1 = relu(h@W1+b1) ; h2 = h1@W2+b2 ; accumulate BN col stats.
#define MLP_ROWS 64
__global__ __launch_bounds__(256)
void gine_mlp(const float* __restrict__ x,
              const float* __restrict__ agg,
              const __bf16* __restrict__ W1,         // [D][D] layer slice (bf16)
              const __bf16* __restrict__ W2,         // [D][D] layer slice (bf16)
              const float* __restrict__ b1,
              const float* __restrict__ b2,
              float* __restrict__ h2out,
              float* __restrict__ colsum,
              float* __restrict__ colsumsq) {
    __shared__ __bf16 sW[D][D + 8];                  // W1 then reused for W2
    __shared__ __bf16 sH[MLP_ROWS][D + 8];           // bf16 intermediate h1
    __shared__ float  sSum[D];
    __shared__ float  sSq[D];

    const int t     = threadIdx.x;
    const int lane  = t & 31;
    const int wave  = t >> 5;
    const int rbase = blockIdx.x * MLP_ROWS;

    const int col16  = lane & 15;
    const int kbase  = (lane < 16) ? 0 : 8;
    const int col    = wave * 16 + col16;
    const int rowoff = (lane < 16) ? 0 : 8;

    if (wave == 0)
        tdm_load_bf16_2d(W1, (unsigned)(size_t)&sW[0][0], D, D);
    if (t < D) { sSum[t] = 0.f; sSq[t] = 0.f; }
    __syncthreads();

    const float bias1 = b1[col];

    // phase 1: GEMM1 + bias + ReLU -> sH (bf16)
    for (int rt = 0; rt < 4; ++rt) {
        const int arow = rbase + rt * 16 + col16;
        v8f c = {};
#pragma unroll
        for (int kk = 0; kk < 4; ++kk) {
            v16bf a, b;
            if (arow < N_NODES) {
                const float* xp = x   + (long long)arow * D + kk * 32;
                const float* ap = agg + (long long)arow * D + kk * 32;
#pragma unroll
                for (int j = 0; j < 8; ++j) {
                    a[j]     = (__bf16)(xp[kbase + j]      + ap[kbase + j]);
                    a[8 + j] = (__bf16)(xp[kbase + 16 + j] + ap[kbase + 16 + j]);
                }
            } else {
#pragma unroll
                for (int j = 0; j < 16; ++j) a[j] = (__bf16)0.f;
            }
#pragma unroll
            for (int j = 0; j < 8; ++j) {
                b[j]     = sW[kk * 32 + kbase + j][col];
                b[8 + j] = sW[kk * 32 + kbase + 16 + j][col];
            }
            c = wmma_bf16(a, b, c);
        }
#pragma unroll
        for (int r = 0; r < 8; ++r)
            sH[rt * 16 + rowoff + r][col] = (__bf16)fmaxf(c[r] + bias1, 0.f);
    }
    __syncthreads();

    if (wave == 0)                                   // overwrite W1 with W2
        tdm_load_bf16_2d(W2, (unsigned)(size_t)&sW[0][0], D, D);
    __syncthreads();

    const float bias2 = b2[col];
    float psum = 0.f, psq = 0.f;

    // phase 2: GEMM2 + bias -> h2, with per-column BN partial sums
    for (int rt = 0; rt < 4; ++rt) {
        const int lrow = rt * 16 + col16;
        v8f c = {};
#pragma unroll
        for (int kk = 0; kk < 4; ++kk) {
            v16bf a, b;
#pragma unroll
            for (int j = 0; j < 8; ++j) {
                a[j]     = sH[lrow][kk * 32 + kbase + j];
                a[8 + j] = sH[lrow][kk * 32 + kbase + 16 + j];
                b[j]     = sW[kk * 32 + kbase + j][col];
                b[8 + j] = sW[kk * 32 + kbase + 16 + j][col];
            }
            c = wmma_bf16(a, b, c);
        }
#pragma unroll
        for (int r = 0; r < 8; ++r) {
            const int grow = rbase + rt * 16 + rowoff + r;
            if (grow < N_NODES) {
                const float v = c[r] + bias2;
                *(float*)((char*)h2out +
                          (size_t)((unsigned)grow * (unsigned)(D * 4) + (unsigned)col * 4u)) = v;
                psum += v;
                psq  += v * v;
            }
        }
    }
    atomicAdd(&sSum[col], psum);                     // ds_add_f32
    atomicAdd(&sSq[col],  psq);
    __syncthreads();
    if (t < D)          atomicAdd(&colsum[t],       sSum[t]);
    else if (t < 2 * D) atomicAdd(&colsumsq[t - D], sSq[t - D]);
}

// ------------------------------------------------------------- batchnorm tail
__global__ void bn_stats(const float* __restrict__ colsum,
                         const float* __restrict__ colsumsq,
                         const float* __restrict__ gamma,
                         const float* __restrict__ beta,
                         float* __restrict__ scale,
                         float* __restrict__ shift) {
    const int n = threadIdx.x;
    if (n < D) {
        const float inv_n = 1.f / (float)N_NODES;
        const float mean  = colsum[n] * inv_n;
        const float var   = colsumsq[n] * inv_n - mean * mean;
        const float inv   = rsqrtf(var + BN_EPS);
        const float sc    = gamma[n] * inv;
        scale[n] = sc;
        shift[n] = beta[n] - mean * sc;
    }
}

__global__ void bn_apply(const float4* __restrict__ h2,
                         const float* __restrict__ scale,
                         const float* __restrict__ shift,
                         float4* __restrict__ xout) {
    const int total4 = N_NODES * (D / 4);
    int i = blockIdx.x * blockDim.x + threadIdx.x;
    const int stride = gridDim.x * blockDim.x;
    for (; i < total4; i += stride) {
        const int cb = (i * 4) & (D - 1);
        const float4 v = h2[i];
        float4 o;
        o.x = fmaxf(v.x * scale[cb + 0] + shift[cb + 0], 0.f);
        o.y = fmaxf(v.y * scale[cb + 1] + shift[cb + 1], 0.f);
        o.z = fmaxf(v.z * scale[cb + 2] + shift[cb + 2], 0.f);
        o.w = fmaxf(v.w * scale[cb + 3] + shift[cb + 3], 0.f);
        xout[i] = o;
    }
}

// --------------------------------------------------------------------- launch
extern "C" void kernel_launch(void* const* d_in, const int* in_sizes, int n_in,
                              void* d_out, int out_size, void* d_ws, size_t ws_size,
                              hipStream_t stream) {
    const float*     x_in  = (const float*)d_in[0];
    const float*     eattr = (const float*)d_in[1];
    const long long* eidx  = (const long long*)d_in[2];
    const float*     Wedge = (const float*)d_in[3];
    const float*     bedge = (const float*)d_in[4];
    const float*     W1    = (const float*)d_in[5];
    const float*     b1    = (const float*)d_in[6];
    const float*     W2    = (const float*)d_in[7];
    const float*     b2    = (const float*)d_in[8];
    const float*     gamma = (const float*)d_in[9];
    const float*     beta  = (const float*)d_in[10];
    float* out = (float*)d_out;

    const size_t ND = (size_t)N_NODES * D;
    char* ws = (char*)d_ws;
    float*  agg      = (float*)ws;  ws += ND * sizeof(float);
    float*  colsum   = (float*)ws;  ws += D * sizeof(float);     // contiguous with agg
    float*  colsumsq = (float*)ws;  ws += D * sizeof(float);     // (zeroed together)
    float*  scale    = (float*)ws;  ws += D * sizeof(float);
    float*  shift    = (float*)ws;  ws += D * sizeof(float);
    float*  h2       = (float*)ws;  ws += ND * sizeof(float);
    float*  xA       = (float*)ws;  ws += ND * sizeof(float);
    float*  xB       = (float*)ws;  ws += ND * sizeof(float);
    __bf16* WedgeBf  = (__bf16*)ws; ws += (size_t)N_LAYERS * EDIM * D * sizeof(__bf16);
    __bf16* W1Bf     = (__bf16*)ws; ws += (size_t)N_LAYERS * D * D * sizeof(__bf16);
    __bf16* W2Bf     = (__bf16*)ws; ws += (size_t)N_LAYERS * D * D * sizeof(__bf16);

    // one-time (per launch) bf16 weight conversion
    f32_to_bf16<<<64,  256, 0, stream>>>(Wedge, WedgeBf, N_LAYERS * EDIM * D);
    f32_to_bf16<<<256, 256, 0, stream>>>(W1,    W1Bf,    N_LAYERS * D * D);
    f32_to_bf16<<<256, 256, 0, stream>>>(W2,    W2Bf,    N_LAYERS * D * D);

    const float* xcur = x_in;
    for (int L = 0; L < N_LAYERS; ++L) {
        zero_f32<<<1024, 256, 0, stream>>>((float4*)agg, (int)((ND + 2 * D) / 4));
        gine_edge<<<N_EDGES / 128, 256, 0, stream>>>(
            xcur, eattr, eidx,
            WedgeBf + (size_t)L * EDIM * D, bedge + L * D, agg);
        gine_mlp<<<(N_NODES + MLP_ROWS - 1) / MLP_ROWS, 256, 0, stream>>>(
            xcur, agg,
            W1Bf + (size_t)L * D * D, W2Bf + (size_t)L * D * D,
            b1 + L * D, b2 + L * D, h2, colsum, colsumsq);
        bn_stats<<<1, 128, 0, stream>>>(colsum, colsumsq,
                                        gamma + L * D, beta + L * D, scale, shift);
        float* xnext = (L == N_LAYERS - 1) ? out : ((L & 1) ? xB : xA);
        bn_apply<<<2048, 256, 0, stream>>>((const float4*)h2, scale, shift, (float4*)xnext);
        xcur = xnext;
    }
}
